// MoELayer_8186207666954
// MI455X (gfx1250) — matrix operand order repaired and checked by reading, not compiled, not run on hardware
//
#include <hip/hip_runtime.h>
#include <hip/hip_bf16.h>
#include <math.h>

// ---------------- problem constants ----------------
#define NTOK   4096      // B*T
#define DMODEL 1024
#define DFF    4096
#define NEXP   8
#define TOPK   2

typedef __attribute__((ext_vector_type(16))) __bf16 v16bf;
typedef __attribute__((ext_vector_type(8)))  __bf16 v8bf;
typedef __attribute__((ext_vector_type(8)))  float  v8f;
typedef __attribute__((ext_vector_type(4)))  unsigned int v4u;
typedef __attribute__((ext_vector_type(8)))  int v8i;
typedef __attribute__((ext_vector_type(4)))  int v4i;

#if __has_builtin(__builtin_amdgcn_tensor_load_to_lds)
#define HAVE_TDM 1
#else
#define HAVE_TDM 0
#endif

// fp32 -> bf16 round-to-nearest-even (bit trick; inf/nan truncate)
__device__ __forceinline__ unsigned short f2bf(float f) {
    union { float f; unsigned u; } v; v.f = f;
    unsigned u = v.u;
    unsigned r = u + 0x7FFFu + ((u >> 16) & 1u);
    if ((u & 0x7F800000u) == 0x7F800000u) r = u;   // inf/nan
    return (unsigned short)(r >> 16);
}

// ---------------- 1) x -> bf16 ----------------
__global__ void convert_x_kernel(const float* __restrict__ x,
                                 unsigned short* __restrict__ xbf, int n4) {
    int i = blockIdx.x * blockDim.x + threadIdx.x;
    int stride = gridDim.x * blockDim.x;
    for (; i < n4; i += stride) {
        float4 v = ((const float4*)x)[i];
        ushort4 o;
        o.x = f2bf(v.x); o.y = f2bf(v.y); o.z = f2bf(v.z); o.w = f2bf(v.w);
        ((ushort4*)xbf)[i] = o;
    }
}

// ---------------- 1b) weights: fp32 (R,C) -> bf16 transposed (C,R), per expert z ----
__global__ void transpose_cvt_kernel(const float* __restrict__ src,
                                     unsigned short* __restrict__ dst,
                                     int R, int C) {
    __shared__ float t[32][33];
    int z = blockIdx.z;
    src += (size_t)z * R * C;
    dst += (size_t)z * R * C;
    int c0 = blockIdx.x * 32, r0 = blockIdx.y * 32;
    int tx = threadIdx.x, ty = threadIdx.y;
#pragma unroll
    for (int i = 0; i < 4; ++i)
        t[ty + i * 8][tx] = src[(size_t)(r0 + ty + i * 8) * C + c0 + tx];
    __syncthreads();
#pragma unroll
    for (int i = 0; i < 4; ++i)
        dst[(size_t)(c0 + ty + i * 8) * R + r0 + tx] = f2bf(t[tx][ty + i * 8]);
}

// ---------------- 2) router: logits, softmax probs, top-2 ----------------
__global__ void router_kernel(const float* __restrict__ x,
                              const float* __restrict__ gw,
                              float* __restrict__ probs,
                              float* __restrict__ topw,
                              int*   __restrict__ topi) {
    int n = blockIdx.x * blockDim.x + threadIdx.x;
    if (n >= NTOK) return;
    float lg[NEXP];
#pragma unroll
    for (int e = 0; e < NEXP; ++e) lg[e] = 0.0f;
    const float* xr = x + (size_t)n * DMODEL;
    for (int d = 0; d < DMODEL; ++d) {
        float xv = xr[d];
        const float* g = gw + (size_t)d * NEXP;
#pragma unroll
        for (int e = 0; e < NEXP; ++e) lg[e] = fmaf(xv, g[e], lg[e]);
    }
    float mx = lg[0];
#pragma unroll
    for (int e = 1; e < NEXP; ++e) mx = fmaxf(mx, lg[e]);
    float s = 0.0f, p[NEXP];
#pragma unroll
    for (int e = 0; e < NEXP; ++e) { p[e] = __expf(lg[e] - mx); s += p[e]; }
    float inv = 1.0f / s;
#pragma unroll
    for (int e = 0; e < NEXP; ++e) probs[(size_t)n * NEXP + e] = p[e] * inv;
    int i0 = 0;
#pragma unroll
    for (int e = 1; e < NEXP; ++e) if (lg[e] > lg[i0]) i0 = e;
    int i1 = -1;
#pragma unroll
    for (int e = 0; e < NEXP; ++e)
        if (e != i0 && (i1 < 0 || lg[e] > lg[i1])) i1 = e;
    float e1 = __expf(lg[i1] - lg[i0]);
    float w0 = 1.0f / (1.0f + e1);
    topw[n * 2 + 0] = w0;
    topw[n * 2 + 1] = e1 * w0;
    topi[n * 2 + 0] = i0;
    topi[n * 2 + 1] = i1;
}

// ---------------- 3) aux loss (fixed-order deterministic reduce) ----------------
__global__ void aux_kernel(const float* __restrict__ probs, float* __restrict__ out_aux) {
    __shared__ float m[NEXP];
    int t = threadIdx.x;
    if (t < NEXP) {
        float s = 0.0f;
        for (int n = 0; n < NTOK; ++n) s += probs[(size_t)n * NEXP + t];
        m[t] = s / (float)NTOK;
    }
    __syncthreads();
    if (t == 0) {
        float a = 0.0f;
        for (int e = 0; e < NEXP; ++e) a += m[e] * m[e];
        *out_aux = a * (float)NEXP;
    }
}

// ---------------- 4) gather lists ----------------
__global__ void zero_counts_kernel(int* counts) {
    if (threadIdx.x < NEXP) counts[threadIdx.x] = 0;
}
__global__ void build_lists_kernel(const int* __restrict__ topi,
                                   int* __restrict__ counts,
                                   int* __restrict__ lists) {
    int n = blockIdx.x * blockDim.x + threadIdx.x;
    if (n >= NTOK) return;
#pragma unroll
    for (int k = 0; k < TOPK; ++k) {
        int e = topi[n * 2 + k];
        int pos = atomicAdd(&counts[e], 1);
        lists[e * NTOK + pos] = (n << 1) | k;   // token id + slot bit
    }
}

// ---------------- 5) expert FFN: gelu(X@W1) @ W2, bf16 WMMA + double-buffered TDM --
#define XSTRIDE 1032    // 1024 + 8 pad halves (row = 2064B, 16B aligned, bank-clean)
#define WSTRIDE 40      // 32 + 8 pad halves  (row = 80B; TDM pad: 16DW + 4DW)
#define HSTRIDE 136     // 128 + 8 pad halves (row = 272B)
#define NSTAGE  ((DFF / 128) * 64)   // 32 chunks x (32 W1 tiles + 32 W2 tiles)

__device__ __forceinline__ v16bf cat8(v8bf a, v8bf b) {
    return __builtin_shufflevector(a, b, 0,1,2,3,4,5,6,7,8,9,10,11,12,13,14,15);
}

// linear stage s -> global source tile (128 rows x 32 halves) and row stride
__device__ __forceinline__ const unsigned short* stage_src(
        int s, int e,
        const unsigned short* __restrict__ w1t,
        const unsigned short* __restrict__ w2t,
        unsigned int* stride) {
    int ffc = s >> 6;            // ff chunk
    int r   = s & 63;
    if (r < 32) {                // phase-1 tile: W1t rows ffc*128.., cols r*32..
        *stride = DMODEL;
        return w1t + ((size_t)e * DFF + ffc * 128) * DMODEL + (r << 5);
    }
    int j = r - 32;              // phase-2 tile: W2t rows dt*128.., cols ffc*128+kk..
    *stride = DFF;
    return w2t + ((size_t)e * DMODEL + (j >> 2) * 128) * DFF + ffc * 128 + ((j & 3) << 5);
}

#if HAVE_TDM
// TDM: load a 128-row x 32-half bf16 tile (64B rows) from global into LDS,
// with hardware padding of 4 DWORDs after every 16 DWORDs -> 80B LDS rows.
__device__ __forceinline__ void tdm_load_tile(const unsigned short* gsrc,
                                              unsigned int rowstride_halves,
                                              unsigned short* lds_dst) {
    unsigned long long ga = (unsigned long long)(uintptr_t)gsrc;
    unsigned int lds_off = (unsigned int)(uintptr_t)lds_dst;   // addr[31:0] = LDS offset
    v4u g0;
    g0.x = 1u;                                              // count=1, user mode
    g0.y = lds_off;                                         // lds_addr
    g0.z = (unsigned int)(ga & 0xFFFFFFFFull);              // global_addr lo
    g0.w = (unsigned int)((ga >> 32) & 0x1FFFFFFu) | (2u << 30);  // hi + type=2
    v8i g1;
    // data_size=2B (1), pad_enable, pad_interval=16DW (3), pad_amount=4DW (3)
    g1[0] = (1 << 16) | (1 << 20) | (3 << 22) | (3 << 25);
    g1[1] = (32 << 16);                   // tensor_dim0[15:0]=32 (bits 63:48)
    g1[2] = (128 << 16);                  // dim0 hi=0, tensor_dim1[15:0]=128
    g1[3] = (32 << 16);                   // dim1 hi=0, tile_dim0=32
    g1[4] = 128;                          // tile_dim1=128, tile_dim2=0
    g1[5] = (int)rowstride_halves;        // tensor_dim0_stride[31:0]
    g1[6] = 0;
    g1[7] = 0;
    v4i gz = {0, 0, 0, 0};
#if defined(__clang_major__) && (__clang_major__ >= 23)
    v8i gz8 = {0, 0, 0, 0, 0, 0, 0, 0};
    __builtin_amdgcn_tensor_load_to_lds(g0, g1, gz, gz, gz8, 0);
#else
    __builtin_amdgcn_tensor_load_to_lds(g0, g1, gz, gz, 0);
#endif
}
#else
// synchronous fallback: 256 threads copy 512 x 16B segments
__device__ __forceinline__ void sync_copy_tile(const unsigned short* __restrict__ gsrc,
                                               unsigned int rowstride_halves,
                                               unsigned short* __restrict__ lds,
                                               int tid) {
#pragma unroll
    for (int j = 0; j < 2; ++j) {
        int i = tid * 2 + j;
        int row = i >> 2, seg = i & 3;
        uint4 v = *(const uint4*)(gsrc + (size_t)row * rowstride_halves + seg * 8);
        *(uint4*)(lds + row * WSTRIDE + seg * 8) = v;
    }
}
#endif

__global__ __launch_bounds__(256) void expert_ffn_kernel(
        const unsigned short* __restrict__ xbf,
        const unsigned short* __restrict__ w1t,   // (E, FF, D) bf16
        const unsigned short* __restrict__ w2t,   // (E, D, FF) bf16
        const int*   __restrict__ counts,
        const int*   __restrict__ lists,
        float*       __restrict__ yslot) {
    __shared__ unsigned short x_lds[16 * XSTRIDE];        // 33024 B
    __shared__ unsigned short wstage[2][128 * WSTRIDE];   // 2 x 10240 B
    __shared__ unsigned short h_lds[16 * HSTRIDE];        //  4352 B
    __shared__ int tok_lds[16];

    const int e    = blockIdx.y;
    const int tile = blockIdx.x;
    const int cnt  = counts[e];
    if (tile * 16 >= cnt) return;            // uniform exit (cnt is uniform)

    const int tid  = threadIdx.x;
    const int wave = tid >> 5;
    const int lane = tid & 31;

    if (tid < 16) {
        int i = tile * 16 + tid;
        tok_lds[tid] = (i < cnt) ? lists[e * NTOK + i] : -1;
    }
    __syncthreads();

    // stage gathered X rows (bf16) into LDS, row-major
    {
        int row = tid >> 4;                  // 0..15
        int col = (tid & 15) * 64;           // 64 halves per thread
        int rec = tok_lds[row];
        uint4* dst = (uint4*)(x_lds + row * XSTRIDE + col);
        if (rec >= 0) {
            const uint4* src = (const uint4*)(xbf + (size_t)(rec >> 1) * DMODEL + col);
#pragma unroll
            for (int j = 0; j < 8; ++j) dst[j] = src[j];
        } else {
            uint4 z = {0u, 0u, 0u, 0u};
#pragma unroll
            for (int j = 0; j < 8; ++j) dst[j] = z;
        }
    }

    // prologue: stage tile 0 into buffer 0
    {
        unsigned int st;
        const unsigned short* g = stage_src(0, e, w1t, w2t, &st);
#if HAVE_TDM
        if (wave == 0) {
            tdm_load_tile(g, st, wstage[0]);
#if __has_builtin(__builtin_amdgcn_s_wait_tensorcnt)
            __builtin_amdgcn_s_wait_tensorcnt(0);
#else
            asm volatile("s_wait_tensorcnt 0x0" ::: "memory");
#endif
        }
#else
        sync_copy_tile(g, st, wstage[0], tid);
#endif
    }
    __syncthreads();

    const int mrow = lane & 15;
    const int kbaA = (lane >> 4) * 8;        // A: K halves {kbaA..+7, kbaA+16..+23}
    const int kbaB = (lane >> 4) * 8;        // B: K halves {kbaB..+7, kbaB+16..+23}
    const int fl   = wave * 16 + (lane & 15);

    v8f acc[8];
    const v8f vz = {0.f,0.f,0.f,0.f,0.f,0.f,0.f,0.f};
#pragma unroll
    for (int dt = 0; dt < 8; ++dt) acc[dt] = vz;

    int s = 0;                               // linear stage counter
    for (int ff0 = 0; ff0 < DFF; ff0 += 128) {
        // ---- phase 1: h(16 x 128) = X(16 x 1024) @ W1[:, ff0:ff0+128] ----
        v8f hacc = vz;
        for (int k0 = 0; k0 < DMODEL; k0 += 32) {
            const unsigned short* wst = wstage[s & 1];
            // issue next tile into the other buffer (overlaps with compute)
            if (s + 1 < NSTAGE) {
                unsigned int st;
                const unsigned short* g = stage_src(s + 1, e, w1t, w2t, &st);
#if HAVE_TDM
                if (wave == 0) tdm_load_tile(g, st, wstage[(s + 1) & 1]);
#else
                sync_copy_tile(g, st, wstage[(s + 1) & 1], tid);
#endif
            }
            v8bf a0 = *(const v8bf*)(x_lds + mrow * XSTRIDE + k0 + kbaA);
            v8bf a1 = *(const v8bf*)(x_lds + mrow * XSTRIDE + k0 + kbaA + 16);
            v8bf b0 = *(const v8bf*)(wst + fl * WSTRIDE + kbaB);
            v8bf b1 = *(const v8bf*)(wst + fl * WSTRIDE + kbaB + 16);
            hacc = __builtin_amdgcn_wmma_f32_16x16x32_bf16(
                false, cat8(a0, a1), false, cat8(b0, b1),
                (short)0, hacc, false, false);
#if HAVE_TDM
            if (wave == 0 && s + 1 < NSTAGE) {
#if __has_builtin(__builtin_amdgcn_s_wait_tensorcnt)
                __builtin_amdgcn_s_wait_tensorcnt(0);
#else
                asm volatile("s_wait_tensorcnt 0x0" ::: "memory");
#endif
            }
#endif
            __syncthreads();                 // next buffer ready; this buffer free
            ++s;
        }
        // exact GELU, write h tile (C layout: lane 0-15 -> M=j, lane 16-31 -> M=8+j)
        {
            int mb = (lane >> 4) * 8;
            int nn = lane & 15;
#pragma unroll
            for (int j = 0; j < 8; ++j) {
                float v = hacc[j];
                float g = 0.5f * v * (1.0f + erff(v * 0.70710678118f));
                h_lds[(mb + j) * HSTRIDE + wave * 16 + nn] = f2bf(g);
            }
        }
        __syncthreads();                     // h visible to all waves
        // ---- phase 2: acc(16 x 1024) += h(16 x 128) @ W2[ff0:ff0+128, :] ----
#pragma unroll
        for (int dt = 0; dt < 8; ++dt) {
#pragma unroll
            for (int kki = 0; kki < 4; ++kki) {
                const int kk = kki * 32;
                const unsigned short* wst = wstage[s & 1];
                if (s + 1 < NSTAGE) {
                    unsigned int st;
                    const unsigned short* g = stage_src(s + 1, e, w1t, w2t, &st);
#if HAVE_TDM
                    if (wave == 0) tdm_load_tile(g, st, wstage[(s + 1) & 1]);
#else
                    sync_copy_tile(g, st, wstage[(s + 1) & 1], tid);
#endif
                }
                v8bf a0 = *(const v8bf*)(h_lds + mrow * HSTRIDE + kk + kbaA);
                v8bf a1 = *(const v8bf*)(h_lds + mrow * HSTRIDE + kk + kbaA + 16);
                v8bf b0 = *(const v8bf*)(wst + fl * WSTRIDE + kbaB);
                v8bf b1 = *(const v8bf*)(wst + fl * WSTRIDE + kbaB + 16);
                acc[dt] = __builtin_amdgcn_wmma_f32_16x16x32_bf16(
                    false, cat8(a0, a1), false, cat8(b0, b1),
                    (short)0, acc[dt], false, false);
#if HAVE_TDM
                if (wave == 0 && s + 1 < NSTAGE) {
#if __has_builtin(__builtin_amdgcn_s_wait_tensorcnt)
                    __builtin_amdgcn_s_wait_tensorcnt(0);
#else
                    asm volatile("s_wait_tensorcnt 0x0" ::: "memory");
#endif
                }
#endif
                __syncthreads();
                ++s;
            }
        }
    }

    // write y into per-(token,slot) buffer (deterministic final combine)
    {
        int mb = (lane >> 4) * 8;
        int nn = lane & 15;
#pragma unroll
        for (int j = 0; j < 8; ++j) {
            int rec = tok_lds[mb + j];
            if (rec < 0) continue;
            float* dst = yslot + (size_t)rec * DMODEL;
#pragma unroll
            for (int dt = 0; dt < 8; ++dt)
                dst[dt * 128 + wave * 16 + nn] = acc[dt][j];
        }
    }
}

// ---------------- 6) combine: out[n] = w0*y[n,0] + w1*y[n,1] ----------------
__global__ void combine_kernel(const float* __restrict__ yslot,
                               const float* __restrict__ topw,
                               float* __restrict__ out) {
    int i = blockIdx.x * blockDim.x + threadIdx.x;   // float4 index
    int total = NTOK * (DMODEL / 4);
    int stride = gridDim.x * blockDim.x;
    for (; i < total; i += stride) {
        int n  = i >> 8;
        int d4 = i & 255;
        float w0 = topw[n * 2 + 0];
        float w1v = topw[n * 2 + 1];
        float4 y0 = ((const float4*)(yslot + (size_t)(n * 2 + 0) * DMODEL))[d4];
        float4 y1 = ((const float4*)(yslot + (size_t)(n * 2 + 1) * DMODEL))[d4];
        float4 o;
        o.x = w0 * y0.x + w1v * y1.x;
        o.y = w0 * y0.y + w1v * y1.y;
        o.z = w0 * y0.z + w1v * y1.z;
        o.w = w0 * y0.w + w1v * y1.w;
        ((float4*)(out + (size_t)n * DMODEL))[d4] = o;
    }
}

// ---------------- launch ----------------
extern "C" void kernel_launch(void* const* d_in, const int* in_sizes, int n_in,
                              void* d_out, int out_size, void* d_ws, size_t ws_size,
                              hipStream_t stream) {
    const float* x      = (const float*)d_in[0];   // (B,T,D)
    const float* gate_w = (const float*)d_in[1];   // (D,E)
    const float* w1     = (const float*)d_in[2];   // (E,D,FF)
    const float* w2     = (const float*)d_in[3];   // (E,FF,D)
    float* out = (float*)d_out;                    // N*D + 1 (aux)

    // workspace carve (256B aligned)
    char* p = (char*)d_ws;
    auto carve = [&](size_t bytes) -> void* {
        void* r = (void*)p;
        p += (bytes + 255) & ~(size_t)255;
        return r;
    };
    unsigned short* xbf  = (unsigned short*)carve((size_t)NTOK * DMODEL * 2);
    float* probs         = (float*)carve((size_t)NTOK * NEXP * 4);
    float* topw          = (float*)carve((size_t)NTOK * TOPK * 4);
    int*   topi          = (int*)carve((size_t)NTOK * TOPK * 4);
    int*   counts        = (int*)carve(64);
    int*   lists         = (int*)carve((size_t)NEXP * NTOK * 4);
    float* yslot         = (float*)carve((size_t)NTOK * TOPK * DMODEL * 4);
    unsigned short* w1t  = (unsigned short*)carve((size_t)NEXP * DFF * DMODEL * 2);
    unsigned short* w2t  = (unsigned short*)carve((size_t)NEXP * DMODEL * DFF * 2);
    (void)ws_size; (void)in_sizes; (void)n_in; (void)out_size;

    convert_x_kernel<<<512, 256, 0, stream>>>(x, xbf, NTOK * DMODEL / 4);
    {
        dim3 tb(32, 8);
        // w1 (E, D=1024, FF=4096) -> w1t (E, FF, D)
        transpose_cvt_kernel<<<dim3(DFF / 32, DMODEL / 32, NEXP), tb, 0, stream>>>(
            w1, w1t, DMODEL, DFF);
        // w2 (E, FF=4096, D=1024) -> w2t (E, D, FF)
        transpose_cvt_kernel<<<dim3(DMODEL / 32, DFF / 32, NEXP), tb, 0, stream>>>(
            w2, w2t, DFF, DMODEL);
    }
    router_kernel<<<NTOK / 256, 256, 0, stream>>>(x, gate_w, probs, topw, topi);
    aux_kernel<<<1, 64, 0, stream>>>(probs, out + (size_t)NTOK * DMODEL);
    zero_counts_kernel<<<1, 32, 0, stream>>>(counts);
    build_lists_kernel<<<NTOK / 256, 256, 0, stream>>>(topi, counts, lists);
    expert_ffn_kernel<<<dim3(256, NEXP), 256, 0, stream>>>(xbf, w1t, w2t, counts, lists, yslot);
    combine_kernel<<<1024, 256, 0, stream>>>(yslot, topw, out);
}
